// DINONewVq_8976481649070
// MI455X (gfx1250) — matrix operand order repaired and checked by reading, not compiled, not run on hardware
//
#include <hip/hip_runtime.h>

typedef __attribute__((ext_vector_type(16))) __bf16 v16bf;
typedef __attribute__((ext_vector_type(8)))  __bf16 v8bf;
typedef __attribute__((ext_vector_type(8)))  float  v8f;

// Problem constants
#define NB    32
#define DD    256
#define HH    32
#define WW    32
#define KK    2048
#define NN    (NB*HH*WW)          // 32768 rows
#define ZQ_ELEMS (NB*DD*HH*WW)    // 8388608

// ---------------------------------------------------------------------------
// Prep 1: codebook f32 -> bf16 (row-major, K x D) + cnorm[k] = sum_d c^2 (f32)
// ---------------------------------------------------------------------------
__global__ __launch_bounds__(256) void vq_prep_cb(const float* __restrict__ cb,
                                                  __bf16* __restrict__ cbf,
                                                  float* __restrict__ cnorm) {
    __shared__ float s[256];
    const int k = blockIdx.x;
    const int d = threadIdx.x;
    const float c = cb[(size_t)k * DD + d];
    cbf[(size_t)k * DD + d] = (__bf16)c;
    s[d] = c * c;
    __syncthreads();
    #pragma unroll
    for (int st = 128; st > 0; st >>= 1) {
        if (d < st) s[d] += s[d + st];
        __syncthreads();
    }
    if (d == 0) cnorm[k] = s[0];
}

// ---------------------------------------------------------------------------
// Prep 2: z (B,D,H,W) f32 -> zbf (N,D) row-major bf16 via LDS transpose,
//         plus znorm[n] = sum_d z^2 in f32.
// ---------------------------------------------------------------------------
__global__ __launch_bounds__(256) void vq_prep_z(const float* __restrict__ z,
                                                 __bf16* __restrict__ zbf,
                                                 float* __restrict__ znorm) {
    __shared__ float tile[64][65];
    __shared__ float sred[256];
    const int tid   = threadIdx.x;
    const int n0    = blockIdx.x * 64;
    const int b     = n0 >> 10;          // 1024 n's per batch image
    const int nbase = n0 & 1023;
    float sq = 0.f;
    for (int by = 0; by < 4; by++) {
        const int d0 = by * 64;
        __syncthreads();
        #pragma unroll
        for (int i = 0; i < 16; i++) {
            const int dd = (tid >> 6) + i * 4;
            const int nn = tid & 63;
            const float v = z[(size_t)b * (DD * HH * WW) + (size_t)(d0 + dd) * (HH * WW) + nbase + nn];
            tile[dd][nn] = v;
            sq += v * v;
        }
        __syncthreads();
        #pragma unroll
        for (int j = 0; j < 16; j++) {
            const int nn2 = (tid >> 6) + j * 4;
            const int dd2 = tid & 63;
            zbf[(size_t)(n0 + nn2) * DD + d0 + dd2] = (__bf16)tile[dd2][nn2];
        }
    }
    sred[tid] = sq;
    __syncthreads();
    if (tid < 64)
        znorm[n0 + tid] = sred[tid] + sred[tid + 64] + sred[tid + 128] + sred[tid + 192];
}

// ---------------------------------------------------------------------------
// Main fused kernel: one block = 16 rows x all 2048 cols.
// 512 threads = 16 waves; wave w owns 8 WMMA column tiles (128 cols).
// Double buffering enforced with sched_barrier: the ~18-load clause for
// K-step kk+1 is pinned above the 8 WMMAs of step kk, so loads are in
// flight for a full iteration and regalloc must keep two B buffers.
// Fused: GEMM -> dist -> argmin -> softmax -> prob store -> z_q -> loss.
// ---------------------------------------------------------------------------
__global__ __launch_bounds__(512) void vq_main(
    const __bf16* __restrict__ zbf,  const __bf16* __restrict__ cbf,
    const float*  __restrict__ znorm, const float* __restrict__ cnorm,
    const float*  __restrict__ zorig, const float* __restrict__ cborig,
    float* __restrict__ out_zq, float* __restrict__ out_prob,
    float* __restrict__ lossp) {

    __shared__ float s_red[16 * 256];
    __shared__ int   s_idx[16 * 256];
    __shared__ float s_rowmin[16];
    __shared__ int   s_rowidx[16];
    __shared__ float s_rowinv[16];

    const int tid   = threadIdx.x;
    const int wave  = tid >> 5;          // 0..15
    const int lane  = tid & 31;
    const int l16   = lane & 15;
    const int khalf = lane >> 4;
    const int row0  = blockIdx.x * 16;
    const int col0  = wave * 128;        // 8 tiles of 16 cols

    // A: row (row0+l16); low lanes K 0-15, high lanes K 16-31 (contiguous)
    const __bf16* aptr  = zbf + (size_t)(row0 + l16) * DD;
    // B: col (col0+l16); low lanes K 0-15, high lanes K 16-31 (contiguous)
    const __bf16* bbase = cbf + (size_t)(col0 + l16) * DD + khalf * 16;

    // ---- prologue: operands for K-step 0
    v16bf Acur, Bcur[8];
    {
        Acur = *(const v16bf*)(aptr + khalf * 16);
        #pragma unroll
        for (int t = 0; t < 8; t++)
            Bcur[t] = *(const v16bf*)(bbase + (size_t)t * 16 * DD);
    }

    v8f acc[8];
    #pragma unroll
    for (int t = 0; t < 8; t++) { v8f zz = {}; acc[t] = zz; }

    // ---- pipelined main loop over K-steps
    #pragma unroll
    for (int kk = 0; kk < 8; kk++) {
        v16bf Anxt, Bnxt[8];
        if (kk < 7) {   // issue next-step loads before this step's WMMAs
            const int kb = (kk + 1) * 32;
            Anxt = *(const v16bf*)(aptr + kb + khalf * 16);
            #pragma unroll
            for (int t = 0; t < 8; t++)
                Bnxt[t] = *(const v16bf*)(bbase + (size_t)t * 16 * DD + kb);
        }
        // Pin the prefetch clause above the WMMA group: loads stay a full
        // iteration ahead, forcing a real double buffer.
        __builtin_amdgcn_sched_barrier(0);
        #pragma unroll
        for (int t = 0; t < 8; t++)
            acc[t] = __builtin_amdgcn_wmma_f32_16x16x32_bf16(
                false, Acur, false, Bcur[t], (short)0, acc[t], false, false);
        __builtin_amdgcn_sched_barrier(0);
        if (kk < 7) {
            Acur = Anxt;
            #pragma unroll
            for (int t = 0; t < 8; t++) Bcur[t] = Bnxt[t];
        }
    }

    // ---- dist = ||z||^2 + ||c||^2 - 2*dot ; per-lane min/argmin over its cols
    v8f zn = *(const v8f*)(znorm + row0 + khalf * 8);   // zn[v] -> row v+8*khalf
    float vmin[8]; int vidx[8];
    #pragma unroll
    for (int v = 0; v < 8; v++) { vmin[v] = 3.4e38f; vidx[v] = 0; }
    #pragma unroll
    for (int t = 0; t < 8; t++) {
        const int col = col0 + t * 16 + l16;
        const float cn = cnorm[col];
        #pragma unroll
        for (int v = 0; v < 8; v++) {
            const float dst = zn[v] + cn - 2.0f * acc[t][v];
            acc[t][v] = dst;
            if (dst < vmin[v]) { vmin[v] = dst; vidx[v] = col; }
        }
    }
    #pragma unroll
    for (int v = 0; v < 8; v++) {
        const int m = v + 8 * khalf;
        const int slot = m * 256 + wave * 16 + l16;
        s_red[slot] = vmin[v];
        s_idx[slot] = vidx[v];
    }
    __syncthreads();
    if (tid < 16) {                       // per-row global min + argmin
        float mn = 3.4e38f; int mi = 0;
        for (int j = 0; j < 256; j++) {
            const float x = s_red[tid * 256 + j];
            if (x < mn) { mn = x; mi = s_idx[tid * 256 + j]; }
        }
        s_rowmin[tid] = mn;
        s_rowidx[tid] = mi;
    }
    __syncthreads();

    // ---- softmax(-dist): e = exp(rowmin - dist); accumulate row sums
    float rm[8];
    #pragma unroll
    for (int v = 0; v < 8; v++) rm[v] = s_rowmin[v + 8 * khalf];
    float vsum[8];
    #pragma unroll
    for (int v = 0; v < 8; v++) vsum[v] = 0.f;
    #pragma unroll
    for (int t = 0; t < 8; t++) {
        #pragma unroll
        for (int v = 0; v < 8; v++) {
            const float e = __expf(rm[v] - acc[t][v]);
            acc[t][v] = e;
            vsum[v] += e;
        }
    }
    #pragma unroll
    for (int v = 0; v < 8; v++) {
        const int m = v + 8 * khalf;
        s_red[m * 256 + wave * 16 + l16] = vsum[v];
    }
    __syncthreads();
    if (tid < 16) {
        float sm = 0.f;
        for (int j = 0; j < 256; j++) sm += s_red[tid * 256 + j];
        s_rowinv[tid] = 1.0f / sm;
    }
    __syncthreads();

    // ---- stream distance_prob (16 contiguous floats per half-wave per store)
    #pragma unroll
    for (int t = 0; t < 8; t++) {
        const size_t colb = (size_t)(col0 + t * 16 + l16);
        #pragma unroll
        for (int v = 0; v < 8; v++) {
            const int m = v + 8 * khalf;
            out_prob[(size_t)(row0 + m) * KK + colb] = acc[t][v] * s_rowinv[m];
        }
    }

    // ---- z_q gather + loss partial (coalesced along contiguous n)
    const int b     = row0 >> 10;
    const int nbase = row0 & 1023;
    const int r     = tid & 15;
    const int dgrp  = tid >> 4;          // 0..31
    const int myidx = s_rowidx[r];
    float lp = 0.f;
    #pragma unroll
    for (int it = 0; it < 8; it++) {
        const int d = it * 32 + dgrp;
        const float c = cborig[(size_t)myidx * DD + d];
        const size_t off = (size_t)b * (DD * HH * WW) + (size_t)d * (HH * WW) + nbase + r;
        const float zv = zorig[off];
        out_zq[off] = c;
        const float df = c - zv;
        lp += df * df;
    }
    __syncthreads();          // s_red reuse
    s_red[tid] = lp;
    __syncthreads();
    #pragma unroll
    for (int st = 256; st > 0; st >>= 1) {
        if (tid < st) s_red[tid] += s_red[tid + st];
        __syncthreads();
    }
    if (tid == 0) lossp[blockIdx.x] = s_red[0];
}

// ---------------------------------------------------------------------------
// Finalize loss: q_loss = (1 + 0.25) * sum(partials) / (N*D)
// ---------------------------------------------------------------------------
__global__ __launch_bounds__(256) void vq_loss(const float* __restrict__ lossp,
                                               float* __restrict__ out_loss) {
    __shared__ float s[256];
    const int tid = threadIdx.x;
    float a = 0.f;
    for (int i = tid; i < 2048; i += 256) a += lossp[i];
    s[tid] = a;
    __syncthreads();
    #pragma unroll
    for (int st = 128; st > 0; st >>= 1) {
        if (tid < st) s[tid] += s[tid + st];
        __syncthreads();
    }
    if (tid == 0) out_loss[0] = 1.25f * s[0] / (float)((double)NN * (double)DD);
}

// ---------------------------------------------------------------------------
extern "C" void kernel_launch(void* const* d_in, const int* in_sizes, int n_in,
                              void* d_out, int out_size, void* d_ws, size_t ws_size,
                              hipStream_t stream) {
    const float* z  = (const float*)d_in[0];   // (32,256,32,32) f32
    const float* cb = (const float*)d_in[1];   // (2048,256) f32
    // d_in[2], d_in[3] = i, it (unused)

    float* out      = (float*)d_out;
    float* out_zq   = out;                     // 8388608 floats
    float* out_loss = out + ZQ_ELEMS;          // 1 float
    float* out_prob = out + ZQ_ELEMS + 1;      // 32768*2048 floats

    char* ws = (char*)d_ws;
    __bf16* zbf  = (__bf16*)(ws);                         // 16 MiB
    __bf16* cbf  = (__bf16*)(ws + 16777216);              //  1 MiB
    float*  znorm = (float*)(ws + 17825792);              // 128 KiB
    float*  cnorm = (float*)(ws + 17956864);              //   8 KiB
    float*  lossp = (float*)(ws + 17965056);              //   8 KiB

    vq_prep_cb<<<KK, 256, 0, stream>>>(cb, cbf, cnorm);
    vq_prep_z <<<NN / 64, 256, 0, stream>>>(z, zbf, znorm);
    vq_main   <<<NN / 16, 512, 0, stream>>>(zbf, cbf, znorm, cnorm, z, cb,
                                            out_zq, out_prob, lossp);
    vq_loss   <<<1, 256, 0, stream>>>(lossp, out_loss);
}
